// Window_Attention_34522947125743
// MI455X (gfx1250) — compile-verified, matrix operand order checked by reading, not run on hardware
//
#include <hip/hip_runtime.h>
#include <hip/hip_bf16.h>

typedef __attribute__((ext_vector_type(16))) __bf16 v16bf;
typedef __attribute__((ext_vector_type(8)))  __bf16 v8bf;
typedef __attribute__((ext_vector_type(8)))  float  v8f;
typedef __attribute__((ext_vector_type(4)))  float  v4f;
typedef __attribute__((ext_vector_type(4)))  int    v4i;

#define HEADS 6
#define HSP   8
#define WSP   8
#define NTOK  64     // tokens per window (8*8)
#define CH    32     // head dim
#define HDIM  192
#define CDIM  192
#define HW    (HDIM*HDIM)
#define NWIN  576    // (192/8)*(192/8)
#define BATCH 8
#define SCALE 0.17677669529663687f  // 32^-0.5

#if defined(__HIP_DEVICE_COMPILE__) && __has_builtin(__builtin_amdgcn_global_load_async_to_lds_b128)
#define HAVE_ASYNC_LDS 1
typedef __attribute__((address_space(1))) v4i* gv4i_p;   // global 16B vector
typedef __attribute__((address_space(3))) v4i* lv4i_p;   // LDS 16B vector
#endif

__device__ __forceinline__ v16bf pack16(v8bf lo, v8bf hi) {
    v16bf r;
#pragma unroll
    for (int i = 0; i < 8; ++i) { r[i] = lo[i]; r[8 + i] = hi[i]; }
    return r;
}

__global__ __launch_bounds__(128)
void win_attn_wmma_bf16(const float* __restrict__ kv,
                        const float* __restrict__ q,
                        const float* __restrict__ mask,
                        float* __restrict__ out)
{
    // LDS staging (~60.5 KB/block; WGP has 320 KB)
    __shared__ __attribute__((aligned(16))) __bf16 sQ [NTOK * CH];   // [n][c]
    __shared__ __attribute__((aligned(16))) __bf16 sK [NTOK * CH];   // [n][c]
    __shared__ __attribute__((aligned(16))) __bf16 sVt[CH * NTOK];   // [c][n]
    __shared__ __attribute__((aligned(16))) float  sS [NTOK * NTOK]; // scores
    __shared__ __attribute__((aligned(16))) float  sM [NTOK * NTOK]; // mask
    __shared__ __attribute__((aligned(16))) __bf16 sP [NTOK * NTOK]; // probs
    __shared__ __attribute__((aligned(16))) float  sO [NTOK * CH];   // out [n][c]
    __shared__ float sRed[128];                                      // softmax partials

    const int tid  = threadIdx.x;
    const int lane = tid & 31;
    const int wv   = tid >> 5;       // wave 0..3 -> 16-row strip
    const int l16  = lane & 15;
    const int half = lane >> 4;      // 0/1
    const int kb   = half * 8;       // per-lane K base for 16-bit A/B fragments

    int wid = blockIdx.x;
    const int head = wid % HEADS;  wid /= HEADS;
    const int b  = wid / NWIN;
    const int uv = wid % NWIN;
    const int u  = uv / (HDIM / HSP);
    const int v  = uv % (HDIM / WSP);

    const size_t plane = (size_t)CDIM * HW;          // elems per batch image
    const float* kp = kv;                            // kv[0]
    const float* vp = kv + (size_t)BATCH * plane;    // kv[1]
    // element offset of (b, head*CH, u*8, v*8)
    const size_t base = ((size_t)b * CDIM + head * CH) * HW
                      + (size_t)(u * HSP) * HDIM + (size_t)(v * WSP);

    // ---------------- mask -> LDS (async DMA when available) -------------
    const float* mrow = mask + (size_t)uv * (NTOK * NTOK);
#ifdef HAVE_ASYNC_LDS
#pragma unroll
    for (int j = 0; j < 8; ++j) {
        int i4 = (tid + j * 128) * 4;                // 1024 float4 total
        __builtin_amdgcn_global_load_async_to_lds_b128(
            (gv4i_p)(mrow + i4), (lv4i_p)(sM + i4), 0, 0);
    }
#else
#pragma unroll
    for (int j = 0; j < 8; ++j) {
        int i4 = tid + j * 128;
        *(v4f*)(sM + i4 * 4) = *(const v4f*)(mrow + i4 * 4);
    }
#endif

    // ---------------- load Q,K,V -> LDS (bf16) ---------------------------
#pragma unroll
    for (int i = 0; i < 2; ++i) {
        int chunk = tid + i * 128;            // 0..255
        int cc = chunk >> 3;                  // channel 0..31
        int h  = chunk & 7;                   // window row 0..7
        size_t g = base + (size_t)cc * HW + (size_t)h * HDIM;
        int nb = h * WSP;                     // token base n = h*8

        v4f q0 = *(const v4f*)(q  + g);
        v4f q1 = *(const v4f*)(q  + g + 4);
        v4f k0 = *(const v4f*)(kp + g);
        v4f k1 = *(const v4f*)(kp + g + 4);
        v4f v0 = *(const v4f*)(vp + g);
        v4f v1 = *(const v4f*)(vp + g + 4);

#pragma unroll
        for (int w = 0; w < 4; ++w) {
            sQ[(nb + w)     * CH + cc] = (__bf16)q0[w];
            sQ[(nb + 4 + w) * CH + cc] = (__bf16)q1[w];
            sK[(nb + w)     * CH + cc] = (__bf16)k0[w];
            sK[(nb + 4 + w) * CH + cc] = (__bf16)k1[w];
        }
        v8bf vb;
#pragma unroll
        for (int w = 0; w < 4; ++w) { vb[w] = (__bf16)v0[w]; vb[4 + w] = (__bf16)v1[w]; }
        *(v8bf*)(sVt + cc * NTOK + nb) = vb;   // V stored transposed [c][n]
    }

#ifdef HAVE_ASYNC_LDS
#if __has_builtin(__builtin_amdgcn_s_wait_asynccnt)
    __builtin_amdgcn_s_wait_asynccnt(0);
#else
    asm volatile("s_wait_asynccnt 0" ::: "memory");
#endif
#endif
    __syncthreads();

    // ---------------- S = Q * K^T  (per-wave 16x64 strip) ----------------
    {
        const int arow = 16 * wv + l16;       // A lane row
        v8bf alo = *(const v8bf*)(sQ + arow * CH + kb);
        v8bf ahi = *(const v8bf*)(sQ + arow * CH + 16 + kb);
        v16bf afrag = pack16(alo, ahi);

        v8f acc[4] = {};
#pragma unroll
        for (int t = 0; t < 4; ++t) {
            int brow = 16 * t + l16;          // column n of S = row of K
            v8bf blo = *(const v8bf*)(sK + brow * CH + kb);
            v8bf bhi = *(const v8bf*)(sK + brow * CH + 16 + kb);
            v16bf bfrag = pack16(blo, bhi);
            acc[t] = __builtin_amdgcn_wmma_f32_16x16x32_bf16(
                false, afrag, false, bfrag, (short)0, acc[t], false, false);
        }
#pragma unroll
        for (int t = 0; t < 4; ++t)
#pragma unroll
            for (int r = 0; r < 8; ++r) {
                int row = 16 * wv + 8 * half + r;
                sS[row * NTOK + 16 * t + l16] = acc[t][r];
            }
    }
    __syncthreads();

    // ------------- softmax: 2 threads per row (scale + mask) -------------
    {
        const int row  = tid >> 1;
        const int col0 = (tid & 1) * 32;
        float* srow = sS + row * NTOK;
        const float* mr = sM + row * NTOK;

        float mx = -3.0e38f;
#pragma unroll 8
        for (int m = col0; m < col0 + 32; ++m) {
            float val = srow[m] * SCALE + mr[m];
            srow[m] = val;
            mx = fmaxf(mx, val);
        }
        sRed[tid] = mx;
        __syncthreads();
        mx = fmaxf(sRed[row * 2], sRed[row * 2 + 1]);

        float sum = 0.0f;
#pragma unroll 8
        for (int m = col0; m < col0 + 32; ++m) {
            float e = __expf(srow[m] - mx);
            srow[m] = e;
            sum += e;
        }
        __syncthreads();               // sRed reuse hazard
        sRed[tid] = sum;
        __syncthreads();
        float inv = 1.0f / (sRed[row * 2] + sRed[row * 2 + 1]);
#pragma unroll 8
        for (int m = col0; m < col0 + 32; ++m)
            sP[row * NTOK + m] = (__bf16)(srow[m] * inv);
    }
    __syncthreads();

    // ---------------- O = P * V (K = 64 -> two chained WMMAs) ------------
    {
        const int prow = 16 * wv + l16;
        v8f acc[2] = {};
#pragma unroll
        for (int kc = 0; kc < 2; ++kc) {
            v8bf plo = *(const v8bf*)(sP + prow * NTOK + 32 * kc + kb);
            v8bf phi = *(const v8bf*)(sP + prow * NTOK + 32 * kc + 16 + kb);
            v16bf pfrag = pack16(plo, phi);
#pragma unroll
            for (int ct = 0; ct < 2; ++ct) {
                int ccol = 16 * ct + l16;     // output channel
                v8bf vlo = *(const v8bf*)(sVt + ccol * NTOK + 32 * kc + kb);
                v8bf vhi = *(const v8bf*)(sVt + ccol * NTOK + 32 * kc + 16 + kb);
                v16bf vfrag = pack16(vlo, vhi);
                acc[ct] = __builtin_amdgcn_wmma_f32_16x16x32_bf16(
                    false, pfrag, false, vfrag, (short)0, acc[ct], false, false);
            }
        }
#pragma unroll
        for (int ct = 0; ct < 2; ++ct)
#pragma unroll
            for (int r = 0; r < 8; ++r) {
                int row = 16 * wv + 8 * half + r;
                sO[row * CH + 16 * ct + l16] = acc[ct][r];
            }
    }
    __syncthreads();

    // ---------------- store O back in (b, d c, u h, v w) layout ----------
#pragma unroll
    for (int i = 0; i < 2; ++i) {
        int chunk = tid + i * 128;
        int cc = chunk >> 3;
        int h  = chunk & 7;
        int nb = h * WSP;
        v4f o0, o1;
#pragma unroll
        for (int w = 0; w < 4; ++w) {
            o0[w] = sO[(nb + w)     * CH + cc];
            o1[w] = sO[(nb + 4 + w) * CH + cc];
        }
        size_t g = base + (size_t)cc * HW + (size_t)h * HDIM;
        *(v4f*)(out + g)     = o0;
        *(v4f*)(out + g + 4) = o1;
    }
}

extern "C" void kernel_launch(void* const* d_in, const int* in_sizes, int n_in,
                              void* d_out, int out_size, void* d_ws, size_t ws_size,
                              hipStream_t stream) {
    (void)in_sizes; (void)n_in; (void)out_size; (void)d_ws; (void)ws_size;
    const float* kv   = (const float*)d_in[0];
    const float* q    = (const float*)d_in[1];
    const float* mask = (const float*)d_in[2];
    float* out = (float*)d_out;

    dim3 grid(BATCH * NWIN * HEADS);   // 27648 (window, head) problems
    dim3 block(128);                   // 4 wave32
    win_attn_wmma_bf16<<<grid, block, 0, stream>>>(kv, q, mask, out);
}
